// PatchMOELayer_73134703116968
// MI455X (gfx1250) — matrix operand order repaired and checked by reading, not compiled, not run on hardware
//
#include <hip/hip_runtime.h>
#include <cmath>

typedef __attribute__((ext_vector_type(16))) _Float16 half16_t;
typedef __attribute__((ext_vector_type(8)))  _Float16 half8_t;
typedef __attribute__((ext_vector_type(8)))  float    float8_t;

#define PS      8
#define NEXP    8
#define CCH     64
#define KTOT    576      // 64 in-channels * 9 taps, k = cin*9 + (ky*3+kx)
#define KHALF   288      // K processed in two halves to fit LDS
#define KHP     296      // padded LDS row stride in halves (592 B: 16B-aligned, bank-friendly)
#define WIN     10
#define NP      9        // patches per dim (72/8)
#define HW      64

__global__ __launch_bounds__(256) void patch_moe_kernel(
    const float* __restrict__ x,      // (32,64,64,64)
    const float* __restrict__ Wg,     // (64,8)
    const float* __restrict__ bg,     // (8)
    const float* __restrict__ Wexp,   // (8,64,64,3,3): row (e,cout) = 576 contiguous floats
    const float* __restrict__ bexp,   // (8,64)
    float* __restrict__ out)          // (32,64,64,64)
{
    __shared__ __align__(16) float    s_win[CCH * WIN * WIN]; // raw window [c][y][x], f32
    __shared__ __align__(16) _Float16 s_b[CCH * KHP];         // im2col half-panel [n][k], f16
    __shared__ float s_feat[CCH];
    __shared__ float s_log[NEXP];
    __shared__ float s_wgt;
    __shared__ int   s_eid;

    const int tid  = threadIdx.x;
    const int lane = tid & 31;
    const int wv   = tid >> 5;

    const int patch = blockIdx.x;
    const int bimg  = patch / (NP * NP);
    const int pr    = patch - bimg * (NP * NP);
    const int pv    = pr / NP;
    const int ph    = pr - pv * NP;

    // ---- Stage 10x10x64 window into LDS via async global->LDS DMA ----------
    // In-range elements: GLOBAL_LOAD_ASYNC_TO_LDS_B32 (ASYNCcnt, no VGPR trip).
    // Halo / padded elements: plain ds_store of 0 (disjoint addresses, DScnt).
    for (int i = tid; i < CCH * WIN * WIN; i += 256) {
        int c  = i / (WIN * WIN);
        int r  = i - c * (WIN * WIN);
        int y  = r / WIN;
        int xx = r - y * WIN;
        int gy = pv * PS + y - 1;
        int gx = ph * PS + xx - 1;
        unsigned lds_off = (unsigned)(uintptr_t)(&s_win[i]);  // low 32 bits = LDS offset
        if ((unsigned)gy < HW && (unsigned)gx < HW) {
            const float* gp = &x[(((size_t)bimg * CCH + c) * HW + gy) * HW + gx];
            unsigned long long ga = (unsigned long long)(uintptr_t)gp;
            asm volatile("global_load_async_to_lds_b32 %0, %1, off"
                         :: "v"(lds_off), "v"(ga) : "memory");
        } else {
            s_win[i] = 0.0f;
        }
    }
    asm volatile("s_wait_asynccnt 0x0" ::: "memory");
    __syncthreads();

    // ---- Gating: mean-pool -> 64x8 GEMV -> softmax -> top-1 ----
    if (tid < CCH) {
        float s = 0.f;
        const float* wp = &s_win[tid * WIN * WIN];
        for (int i = 0; i < WIN * WIN; ++i) s += wp[i];
        s_feat[tid] = s * (1.0f / (WIN * WIN));
    }
    __syncthreads();
    if (tid < NEXP) {
        float l = bg[tid];
        for (int c = 0; c < CCH; ++c) l += s_feat[c] * Wg[c * NEXP + tid];
        s_log[tid] = l;
    }
    __syncthreads();
    if (tid == 0) {
        int am = 0; float mx = s_log[0];
        for (int e = 1; e < NEXP; ++e) { float le = s_log[e]; if (le > mx) { mx = le; am = e; } }
        float den = 0.f;
        for (int e = 0; e < NEXP; ++e) den += __expf(s_log[e] - mx);
        s_wgt = 1.0f / den;   // prob of the argmax expert
        s_eid = am;
    }
    __syncthreads();
    const int   eid = s_eid;
    const float wgt = s_wgt;

    // ---- Implicit GEMM: D[64 cout x 64 pos] = Wexp[e] (64x576) * im2col (576x64) ----
    // 16 tiles of 16x16; 8 waves -> each wave owns 1 M-block x 2 N-blocks.
    const int mblock = wv & 3;
    const int nb0    = (wv >> 2) * 2;

    const int mrowA = mblock * 16 + (lane & 15);          // A: lane -> M row
    const int b8    = (lane >> 4) << 3;                   // A K sub-base: 0 / 8
    const int kb    = (lane >> 4) << 4;                   // unused K sub-base: 0 / 16 (B uses b8)
    const float* arow = Wexp + ((size_t)eid * CCH + mrowA) * KTOT;
    __builtin_prefetch(arow, 0, 1);

    const int n0  = nb0 * 16 + (lane & 15);
    const int n1  = n0 + 16;

    float8_t acc0 = {};
    float8_t acc1 = {};

    for (int hh = 0; hh < 2; ++hh) {
        // -- Cooperative im2col of this K-half into LDS (f16, [n][k] row-major) --
        __syncthreads();   // previous half's fragments fully consumed
        for (int i = tid; i < CCH * KHALF; i += 256) {
            int n  = i / KHALF;
            int kl = i - n * KHALF;
            int kg = hh * KHALF + kl;
            int c  = kg / 9;
            int tp = kg - c * 9;
            int ky = tp / 3;
            int kx = tp - ky * 3;
            int py = n >> 3, px = n & 7;
            s_b[n * KHP + kl] =
                (_Float16)s_win[c * (WIN * WIN) + (py + ky) * WIN + (px + kx)];
        }
        __syncthreads();

        // -- 9 K-steps of 32 over this half --
#pragma unroll
        for (int kk = 0; kk < KHALF / 32; ++kk) {
            const int k0 = hh * KHALF + kk * 32;
            // A fragment: f32 global (contiguous, L2-resident) -> f16
            half16_t a;
            {
                const float4 c0 = *(const float4*)(arow + k0 + b8);
                const float4 c1 = *(const float4*)(arow + k0 + b8 + 4);
                const float4 c2 = *(const float4*)(arow + k0 + 16 + b8);
                const float4 c3 = *(const float4*)(arow + k0 + 16 + b8 + 4);
                a[0]=(_Float16)c0.x;  a[1]=(_Float16)c0.y;  a[2]=(_Float16)c0.z;  a[3]=(_Float16)c0.w;
                a[4]=(_Float16)c1.x;  a[5]=(_Float16)c1.y;  a[6]=(_Float16)c1.z;  a[7]=(_Float16)c1.w;
                a[8]=(_Float16)c2.x;  a[9]=(_Float16)c2.y;  a[10]=(_Float16)c2.z; a[11]=(_Float16)c2.w;
                a[12]=(_Float16)c3.x; a[13]=(_Float16)c3.y; a[14]=(_Float16)c3.z; a[15]=(_Float16)c3.w;
            }
            // B fragments: two ds_load_b128 each (k 8h..8h+7 and 16+8h..16+8h+7 per lane)
            const _Float16* bp0 = &s_b[n0 * KHP + kk * 32 + b8];
            const _Float16* bp1 = &s_b[n1 * KHP + kk * 32 + b8];
            half8_t lo0 = *(const half8_t*)bp0;
            half8_t hi0 = *(const half8_t*)(bp0 + 16);
            half8_t lo1 = *(const half8_t*)bp1;
            half8_t hi1 = *(const half8_t*)(bp1 + 16);
            half16_t bf0 = __builtin_shufflevector(lo0, hi0, 0,1,2,3,4,5,6,7,8,9,10,11,12,13,14,15);
            half16_t bf1 = __builtin_shufflevector(lo1, hi1, 0,1,2,3,4,5,6,7,8,9,10,11,12,13,14,15);

            acc0 = __builtin_amdgcn_wmma_f32_16x16x32_f16(false, a, false, bf0, (short)0, acc0, false, false);
            acc1 = __builtin_amdgcn_wmma_f32_16x16x32_f16(false, a, false, bf1, (short)0, acc1, false, false);
        }
    }

    // ---- Epilogue: bias + ReLU + gate-weight; crop checks hoisted per lane ----
    const int mhi  = (lane >> 4) << 3;   // D: lanes 16-31 hold M+8
    const int py0 = n0 >> 3, px0 = n0 & 7;
    const int py1 = n1 >> 3, px1 = n1 & 7;
    const int gy0 = pv * PS + py0, gx0 = ph * PS + px0;
    const int gy1 = pv * PS + py1, gx1 = ph * PS + px1;
    const float* brow = bexp + eid * CCH;
    float* obase = out + (size_t)bimg * CCH * HW * HW;

    if (gy0 < HW && gx0 < HW) {
        float* o0 = obase + gy0 * HW + gx0;
#pragma unroll
        for (int r = 0; r < 8; ++r) {
            const int mo = mblock * 16 + mhi + r;
            float v = fmaxf(acc0[r] + brow[mo], 0.f) * wgt;
            o0[(size_t)mo * (HW * HW)] = v;
        }
    }
    if (gy1 < HW && gx1 < HW) {
        float* o1 = obase + gy1 * HW + gx1;
#pragma unroll
        for (int r = 0; r < 8; ++r) {
            const int mo = mblock * 16 + mhi + r;
            float v = fmaxf(acc1[r] + brow[mo], 0.f) * wgt;
            o1[(size_t)mo * (HW * HW)] = v;
        }
    }
}

extern "C" void kernel_launch(void* const* d_in, const int* in_sizes, int n_in,
                              void* d_out, int out_size, void* d_ws, size_t ws_size,
                              hipStream_t stream) {
    (void)in_sizes; (void)n_in; (void)d_ws; (void)ws_size; (void)out_size;
    const float* x    = (const float*)d_in[0];
    const float* Wg   = (const float*)d_in[1];
    const float* bg   = (const float*)d_in[2];
    const float* Wexp = (const float*)d_in[3];
    const float* bexp = (const float*)d_in[4];
    float* out = (float*)d_out;

    dim3 grid(32 * NP * NP);   // one block per patch: 2592
    dim3 block(256);           // 8 wave32
    patch_moe_kernel<<<grid, block, 0, stream>>>(x, Wg, bg, Wexp, bexp, out);
}
